// GCN_22728966930472
// MI455X (gfx1250) — matrix operand order repaired and checked by reading, not compiled, not run on hardware
//
#include <hip/hip_runtime.h>

#define N_NODES 50000
#define N_EDGES 600000
#define HDIM    128
#define NGRAPH  64
#define NPAD    50048   // 391 * 128 (rows padded to 128 for WMMA tiling)
#define MBLK    391

typedef __bf16 bf16_t;
typedef __attribute__((ext_vector_type(16))) __bf16 v16bf;
typedef __attribute__((ext_vector_type(8)))  __bf16 v8bf;
typedef __attribute__((ext_vector_type(8)))  float  v8f;

// ---------------- workspace layout ----------------
static constexpr size_t alignup(size_t x) { return (x + 255) & ~(size_t)255; }
static constexpr size_t OFF_DINV = 0;
static constexpr size_t OFF_NORM = alignup(OFF_DINV + (size_t)N_NODES * 4);
static constexpr size_t OFF_A    = alignup(OFF_NORM + (size_t)N_EDGES * 4);   // matmul output h
static constexpr size_t OFF_B    = alignup(OFF_A + (size_t)NPAD * HDIM * 4);  // aggregate / activation
static constexpr size_t OFF_POOL = alignup(OFF_B + (size_t)NPAD * HDIM * 4);
static constexpr size_t OFF_CNT  = alignup(OFF_POOL + (size_t)NGRAPH * HDIM * 4);

// ---------------- degree / norm ----------------
__global__ void k_deg_init(float* __restrict__ deg) {
  int i = blockIdx.x * 256 + threadIdx.x;
  if (i < N_NODES) deg[i] = 1.0f;   // self-loop contribution
}

__global__ void k_deg_accum(const int* __restrict__ dst, float* __restrict__ deg) {
  int e = blockIdx.x * 256 + threadIdx.x;
  if (e < N_EDGES) atomicAdd(&deg[dst[e]], 1.0f);
}

__global__ void k_rsqrt(float* __restrict__ deg) {
  int i = blockIdx.x * 256 + threadIdx.x;
  if (i < N_NODES) deg[i] = rsqrtf(deg[i]);
}

__global__ void k_norm(const int* __restrict__ src, const int* __restrict__ dst,
                       const float* __restrict__ dinv, float* __restrict__ norm) {
  int e = blockIdx.x * 256 + threadIdx.x;
  if (e < N_EDGES) norm[e] = dinv[src[e]] * dinv[dst[e]];
}

// ---------------- generic zero fill ----------------
__global__ void k_zero(float* __restrict__ p, int n) {
  int t = blockIdx.x * 256 + threadIdx.x;
  if (t < n) p[t] = 0.0f;
}

// ---------------- layer 1: x[N,3] @ W1[3,128] ----------------
__global__ void k_lin1(const float* __restrict__ x, const float* __restrict__ W1,
                       float* __restrict__ out) {
  int t = blockIdx.x * 256 + threadIdx.x;
  if (t >= N_NODES * HDIM) return;
  int n = t >> 7, f = t & (HDIM - 1);
  float acc = x[n * 3 + 0] * W1[0 * HDIM + f];
  acc = fmaf(x[n * 3 + 1], W1[1 * HDIM + f], acc);
  acc = fmaf(x[n * 3 + 2], W1[2 * HDIM + f], acc);
  out[t] = acc;
}

// ---------------- WMMA GEMM: C[NPAD,128] = A_f32[NPAD,128] @ W_f32[128,128] in bf16 ----------------
#define WT_STRIDE 136   // padded bf16 stride (272 B) to spread LDS banks

__launch_bounds__(256)
__global__ void k_gemm_bf16(const float* __restrict__ A,
                            const float* __restrict__ W,
                            float* __restrict__ C) {
  __shared__ __align__(16) bf16_t ldsWT[HDIM * WT_STRIDE];
  const int tid = threadIdx.x;

  // Stage W^T into LDS as bf16: ldsWT[n*WT_STRIDE + k] = bf16(W[k,n])
  for (int i = tid; i < HDIM * HDIM; i += 256) {
    const int k = i >> 7;
    const int n = i & (HDIM - 1);
    ldsWT[n * WT_STRIDE + k] = (bf16_t)W[i];
  }

  const int wave  = tid >> 5;
  const int lane  = tid & 31;
  const int l16   = lane & 15;
  const int lhalf = lane >> 4;                  // 0: lanes 0-15, 1: lanes 16-31
  const int rowBase = blockIdx.x * 128 + wave * 16;
  const float* Arow = A + (size_t)(rowBase + l16) * HDIM;
  __builtin_prefetch(Arow, 0, 1);               // global_prefetch_b8

  __syncthreads();

  v8f acc[8] = {};
  #pragma unroll
  for (int ks = 0; ks < 4; ++ks) {
    // 16-bit A 16x32 layout: lanes 0-15 hold K {0..7,16..23}; lanes 16-31 hold K {8..15,24..31}
    const int ka = ks * 32 + lhalf * 8;
    v8f f0 = *(const v8f*)(Arow + ka);
    v8f f1 = *(const v8f*)(Arow + ka + 16);
    v8bf a0 = __builtin_convertvector(f0, v8bf);
    v8bf a1 = __builtin_convertvector(f1, v8bf);
    v16bf a = __builtin_shufflevector(a0, a1, 0,1,2,3,4,5,6,7,8,9,10,11,12,13,14,15);

    // 16-bit B 32x16 layout: lane = column, lanes 0-15 K {0..7,16..23}; lanes 16-31 K {8..15,24..31}
    const int kb = ks * 32 + lhalf * 8;
    #pragma unroll
    for (int c = 0; c < 8; ++c) {
      const bf16_t* Bp = &ldsWT[(c * 16 + l16) * WT_STRIDE + kb];
      v8bf b0 = *(const v8bf*)(Bp);
      v8bf b1 = *(const v8bf*)(Bp + 16);
      v16bf b = __builtin_shufflevector(b0, b1, 0,1,2,3,4,5,6,7,8,9,10,11,12,13,14,15);
      acc[c] = __builtin_amdgcn_wmma_f32_16x16x32_bf16(
          /*neg_a=*/false, a, /*neg_b=*/false, b,
          /*c_mod=*/(short)0, acc[c], /*reuse_a=*/false, /*reuse_b=*/false);
    }
  }

  // D layout: VGPR r -> row r + lhalf*8; lane -> column l16
  #pragma unroll
  for (int c = 0; c < 8; ++c) {
    const int col = c * 16 + l16;
    #pragma unroll
    for (int r = 0; r < 8; ++r) {
      const int row = rowBase + lhalf * 8 + r;
      if (row < N_NODES) C[(size_t)row * HDIM + col] = acc[c][r];
    }
  }
}

// ---------------- edge scatter: out[dst] += h[src] * norm ----------------
__global__ void k_scatter(const int* __restrict__ src, const int* __restrict__ dst,
                          const float* __restrict__ norm, const float* __restrict__ h,
                          float* __restrict__ out) {
  const int e = blockIdx.x * 8 + (threadIdx.x >> 5);
  if (e >= N_EDGES) return;
  const int lane = threadIdx.x & 31;
  const int s = src[e], d = dst[e];
  const float w = norm[e];
  const float4 v = *(const float4*)(h + (size_t)s * HDIM + lane * 4);
  float* op = out + (size_t)d * HDIM + lane * 4;
  atomicAdd(op + 0, v.x * w);
  atomicAdd(op + 1, v.y * w);
  atomicAdd(op + 2, v.z * w);
  atomicAdd(op + 3, v.w * w);
}

// ---------------- epilogue: self-loop + bias + ReLU (in-place on agg) ----------------
__global__ void k_bias_relu(float* __restrict__ agg, const float* __restrict__ h,
                            const float* __restrict__ dinv, const float* __restrict__ b) {
  int t = blockIdx.x * 256 + threadIdx.x;
  if (t >= N_NODES * HDIM) return;
  int n = t >> 7, f = t & (HDIM - 1);
  float di = dinv[n];
  float v = agg[t] + h[t] * di * di + b[f];
  agg[t] = v > 0.0f ? v : 0.0f;
}

// ---------------- global mean pool (atomics) ----------------
__global__ void k_pool(const float* __restrict__ act, const int* __restrict__ batch,
                       float* __restrict__ pooled, float* __restrict__ cnt) {
  int t = blockIdx.x * 256 + threadIdx.x;
  if (t >= N_NODES * HDIM) return;
  int n = t >> 7, f = t & (HDIM - 1);
  int g = batch[n];
  atomicAdd(&pooled[(size_t)g * HDIM + f], act[t]);
  if (f == 0) atomicAdd(&cnt[g], 1.0f);
}

// ---------------- output head: (pooled/cnt) @ Wout + bout ----------------
__global__ void k_out(const float* __restrict__ pooled, const float* __restrict__ cnt,
                      const float* __restrict__ Wout, const float* __restrict__ bout,
                      float* __restrict__ out) {
  int t = blockIdx.x * 256 + threadIdx.x;
  if (t >= NGRAPH * HDIM) return;
  int g = t >> 7, o = t & (HDIM - 1);
  float inv = 1.0f / fmaxf(cnt[g], 1.0f);
  const float* pg = pooled + (size_t)g * HDIM;
  float acc = bout[o];
  #pragma unroll 8
  for (int k = 0; k < HDIM; ++k) acc = fmaf(pg[k] * inv, Wout[k * HDIM + o], acc);
  out[t] = acc;
}

// ---------------- host launcher ----------------
extern "C" void kernel_launch(void* const* d_in, const int* in_sizes, int n_in,
                              void* d_out, int out_size, void* d_ws, size_t ws_size,
                              hipStream_t stream) {
  (void)in_sizes; (void)n_in; (void)out_size; (void)ws_size;

  const float* x     = (const float*)d_in[0];
  const int*   ei    = (const int*)d_in[1];
  const int*   batch = (const int*)d_in[2];
  const float* W1    = (const float*)d_in[3];
  const float* b1    = (const float*)d_in[4];
  const float* W2    = (const float*)d_in[5];
  const float* b2    = (const float*)d_in[6];
  const float* W3    = (const float*)d_in[7];
  const float* b3    = (const float*)d_in[8];
  const float* Wout  = (const float*)d_in[9];
  const float* bout  = (const float*)d_in[10];
  float* out = (float*)d_out;

  const int* src = ei;             // edge_index[0]
  const int* dst = ei + N_EDGES;   // edge_index[1]

  char* ws = (char*)d_ws;
  float* dinv   = (float*)(ws + OFF_DINV);
  float* norm   = (float*)(ws + OFF_NORM);
  float* bufA   = (float*)(ws + OFF_A);     // per-layer matmul output h
  float* bufB   = (float*)(ws + OFF_B);     // aggregation target / activations
  float* pooled = (float*)(ws + OFF_POOL);
  float* cnt    = (float*)(ws + OFF_CNT);

  const int gN  = (N_NODES + 255) / 256;
  const int gE  = (N_EDGES + 255) / 256;
  const int gNH = (N_NODES * HDIM + 255) / 256;
  const int gPH = (NPAD * HDIM) / 256;
  const int gSC = (N_EDGES + 7) / 8;

  // degree / symmetric norm
  k_deg_init <<<gN, 256, 0, stream>>>(dinv);
  k_deg_accum<<<gE, 256, 0, stream>>>(dst, dinv);
  k_rsqrt    <<<gN, 256, 0, stream>>>(dinv);
  k_norm     <<<gE, 256, 0, stream>>>(src, dst, dinv, norm);

  // ---- layer 1 (F_IN=3, plain VALU matmul) ----
  k_lin1     <<<gNH, 256, 0, stream>>>(x, W1, bufA);
  k_zero     <<<gPH, 256, 0, stream>>>(bufB, NPAD * HDIM);
  k_scatter  <<<gSC, 256, 0, stream>>>(src, dst, norm, bufA, bufB);
  k_bias_relu<<<gNH, 256, 0, stream>>>(bufB, bufA, dinv, b1);

  // ---- layer 2 (bf16 WMMA GEMM) ----
  k_gemm_bf16<<<MBLK, 256, 0, stream>>>(bufB, W2, bufA);
  k_zero     <<<gPH, 256, 0, stream>>>(bufB, NPAD * HDIM);
  k_scatter  <<<gSC, 256, 0, stream>>>(src, dst, norm, bufA, bufB);
  k_bias_relu<<<gNH, 256, 0, stream>>>(bufB, bufA, dinv, b2);

  // ---- layer 3 (bf16 WMMA GEMM) ----
  k_gemm_bf16<<<MBLK, 256, 0, stream>>>(bufB, W3, bufA);
  k_zero     <<<gPH, 256, 0, stream>>>(bufB, NPAD * HDIM);
  k_scatter  <<<gSC, 256, 0, stream>>>(src, dst, norm, bufA, bufB);
  k_bias_relu<<<gNH, 256, 0, stream>>>(bufB, bufA, dinv, b3);

  // ---- global mean pool + output head ----
  k_zero<<<(NGRAPH * HDIM + 255) / 256, 256, 0, stream>>>(pooled, NGRAPH * HDIM);
  k_zero<<<1, 256, 0, stream>>>(cnt, NGRAPH);
  k_pool<<<gNH, 256, 0, stream>>>(bufB, batch, pooled, cnt);
  k_out <<<(NGRAPH * HDIM + 255) / 256, 256, 0, stream>>>(pooled, cnt, Wout, bout, out);
}
